// AutoregressiveDecoder_77764677862225
// MI455X (gfx1250) — compile-verified, matrix operand order checked by reading, not run on hardware
//
#include <hip/hip_runtime.h>

typedef __attribute__((ext_vector_type(16))) __bf16 v16bf;
typedef __attribute__((ext_vector_type(8)))  float  v8f;
typedef __attribute__((ext_vector_type(8)))  int    v8i;

#define B_    256
#define T_    2048
#define FIN   62
#define H_    256
#define OUT_  2
#define NOUT  8192
#define KIN   64      // FIN + 2 (x_t ++ prev_out)
#define G4H   1024    // 4*H gate columns

// workspace layout (bytes); fragments are 1KB each (32 lanes x 32B)
#define WS_WHH  0u
#define WS_WIH  (WS_WHH + 64u * 8u * 1024u)   // W_hh: 64 n-tiles x 8 k-tiles
#define WS_FC   (WS_WIH + 64u * 2u * 1024u)   // W_ih: 64 n-tiles x 2 k-tiles
#define WS_BIAS (WS_FC  +  8u * 1024u)        // fc:    1 n-tile x 8 k-tiles

__device__ __forceinline__ unsigned short f2bf(float f) {
  unsigned u = __builtin_bit_cast(unsigned, f);
  unsigned r = (u + 0x7FFFu + ((u >> 16) & 1u)) >> 16;   // round-to-nearest-even
  return (unsigned short)r;
}
__device__ __forceinline__ float sigm(float x) { return 1.f / (1.f + __expf(-x)); }
__device__ __forceinline__ float tanh_(float x) { return 2.f / (1.f + __expf(-2.f * x)) - 1.f; }
__device__ __forceinline__ v8f splat8(float s) { v8f v = {s,s,s,s,s,s,s,s}; return v; }

// A-operand (16xK, 16-bit) fragment gather from LDS, per ISA VGPR layout:
// lane L: m = L%16, half = L/16; dword v holds K = (v/4)*16 + half*8 + (v%4)*2 (+0,+1)
template <int STRIDE>
__device__ __forceinline__ v16bf load_a_sh(unsigned short (*S)[STRIDE], int m, int kbase, int hf) {
  int ai[8];
#pragma unroll
  for (int v = 0; v < 8; ++v) {
    int k = kbase + ((v >> 2) << 4) + (hf << 3) + ((v & 3) << 1);
    ai[v] = *(const int*)&S[m][k];
  }
  v8i tmp = {ai[0], ai[1], ai[2], ai[3], ai[4], ai[5], ai[6], ai[7]};
  return __builtin_bit_cast(v16bf, tmp);
}

// ---------------- prep kernels ----------------

__global__ void add_bias_kernel(const float* __restrict__ bih, const float* __restrict__ bhh,
                                float* __restrict__ bias) {
  int i = blockIdx.x * blockDim.x + threadIdx.x;
  if (i < G4H) bias[i] = bih[i] + bhh[i];
}

// Pre-swizzle W (n_rows x K, row-major f32) into per-lane WMMA B-operand bf16
// fragments: fragment f = ntile*kt_count + kt occupies 1KB; lane L owns bytes
// [L*32, L*32+32).
__global__ void swizzle_kernel(const float* __restrict__ W, unsigned short* __restrict__ dst,
                               int K, int n_valid, int kt_count) {
  int frag = blockIdx.x;
  int ntile = frag / kt_count, kt = frag - ntile * kt_count;
  int lane = threadIdx.x;
  int hf = lane >> 4, nl = lane & 15;
  int n = ntile * 16 + nl;
#pragma unroll
  for (int e = 0; e < 16; ++e) {
    int v = e >> 1, p = e & 1;
    int k = kt * 32 + ((v >> 2) << 4) + (hf << 3) + ((v & 3) << 1) + p;
    float val = (n < n_valid) ? W[n * K + k] : 0.f;
    dst[frag * 512 + lane * 16 + e] = f2bf(val);
  }
}

// ---------------- persistent LSTM kernel ----------------
// grid = 16 (batch tiles of 16 rows), block = 256 (8 waves).
// wave w owns hidden columns [w*32, w*32+32) of all 4 gates -> cell state c
// lives in this wave's registers in WMMA C/D layout; h exchanged via LDS bf16.
// Weights stream from (permanently hot) L2 every step. A laundered 32-bit
// offset (not a laundered pointer!) keeps the loads inside the t-loop without
// blocking address-space inference, so they lower to global_load_b128 (saddr)
// tracked by LOADcnt only -- not flat_load coupled to DScnt.
__global__ __launch_bounds__(256) void lstm_kernel(
    const float* __restrict__ x, const unsigned short* __restrict__ whh,
    const unsigned short* __restrict__ wih, const unsigned short* __restrict__ wfc,
    const float* __restrict__ bias, const float* __restrict__ fcb,
    float* __restrict__ mtv) {
  __shared__ unsigned short inp_s[16][KIN + 2];  // bf16 bits, padded rows
  __shared__ unsigned short h_s[16][H_ + 2];     // bf16 bits, padded rows

  const int tid = threadIdx.x;
  const int lane = tid & 31;
  const int w = tid >> 5;   // wave id 0..7
  const int hf = lane >> 4; // lane half
  const int nl = lane & 15;
  const int bb = blockIdx.x * 16;  // batch base

  for (int i = tid; i < 16 * (KIN + 2); i += 256) ((unsigned short*)inp_s)[i] = 0;
  for (int i = tid; i < 16 * (H_ + 2);  i += 256) ((unsigned short*)h_s)[i]  = 0;

  // per-accumulator bias (depends only on gate column = lane)
  float bias_r[8];
#pragma unroll
  for (int a = 0; a < 8; ++a) {
    int ntile = (a >> 1) * 16 + w * 2 + (a & 1);  // gate G = a>>1, half = a&1
    bias_r[a] = bias[ntile * 16 + nl];
  }
  float fcb_l = (nl < OUT_) ? fcb[nl] : 0.f;

  v8f cst[2];  // cell state, C/D layout, one tile per 16-col half
  cst[0] = splat8(0.f);
  cst[1] = splat8(0.f);

  __syncthreads();

  for (int t = 0; t < T_; ++t) {
    // Launder a uniform zero offset each iteration: defeats LICM (no 640-VGPR
    // hoist + scratch spill) while keeping the weight pointers' global
    // address space intact (global_load, not flat_load).
    unsigned toff = 0;
    asm volatile("" : "+s"(toff));
    const unsigned short* whh_t = whh + toff;
    const unsigned short* wih_t = wih + toff;
    const unsigned short* wfc_t = wfc + toff;

    // stage x_t (cols 0..61) into LDS as bf16; cols 62/63 hold prev fc output
    for (int i = tid; i < 16 * FIN; i += 256) {
      int m = i / FIN, k = i - m * FIN;
      inp_s[m][k] = f2bf(x[((long)(bb + m) * T_ + t) * FIN + k]);
    }
    __syncthreads();

    v8f acc[8];
#pragma unroll
    for (int a = 0; a < 8; ++a) acc[a] = splat8(bias_r[a]);

    // input contribution: inp[16,64] @ W_ih^T  (2 K-tiles)
#pragma unroll
    for (int kt = 0; kt < 2; ++kt) {
      v16bf afr = load_a_sh<KIN + 2>(inp_s, nl, kt * 32, hf);
#pragma unroll
      for (int a = 0; a < 8; ++a) {
        int ntile = (a >> 1) * 16 + w * 2 + (a & 1);
        v8i bfrag = *(const v8i*)(wih_t + (ntile * 2 + kt) * 512 + lane * 16);
        acc[a] = __builtin_amdgcn_wmma_f32_16x16x32_bf16(
            false, afr, false, __builtin_bit_cast(v16bf, bfrag),
            (short)0, acc[a], false, false);
      }
    }
    // recurrent contribution: h[16,256] @ W_hh^T  (8 K-tiles)
#pragma unroll
    for (int kt = 0; kt < 8; ++kt) {
      v16bf afr = load_a_sh<H_ + 2>(h_s, nl, kt * 32, hf);
#pragma unroll
      for (int a = 0; a < 8; ++a) {
        int ntile = (a >> 1) * 16 + w * 2 + (a & 1);
        v8i bfrag = *(const v8i*)(whh_t + (ntile * 8 + kt) * 512 + lane * 16);
        acc[a] = __builtin_amdgcn_wmma_f32_16x16x32_bf16(
            false, afr, false, __builtin_bit_cast(v16bf, bfrag),
            (short)0, acc[a], false, false);
      }
    }

    __syncthreads();  // all h_s reads done before overwrite

    // LSTM cell update, fully in-register; write new h (bf16) to LDS
#pragma unroll
    for (int half = 0; half < 2; ++half) {
      v8f ig = acc[0 + half], fg = acc[2 + half], gg = acc[4 + half], og = acc[6 + half];
      int col = w * 32 + half * 16 + nl;
#pragma unroll
      for (int r = 0; r < 8; ++r) {
        float i_ = sigm(ig[r]);
        float f_ = sigm(fg[r]);
        float g_ = tanh_(gg[r]);
        float o_ = sigm(og[r]);
        float c_ = f_ * cst[half][r] + i_ * g_;
        cst[half][r] = c_;
        float h_ = o_ * tanh_(c_);
        h_s[r + 8 * hf][col] = f2bf(h_);
      }
    }
    __syncthreads();

    // fc head on wave 0: out[16,2] = h @ fc_W^T + fc_b  (one padded N-tile)
    if (w == 0) {
      v8f facc = splat8(fcb_l);
#pragma unroll
      for (int kt = 0; kt < 8; ++kt) {
        v16bf afr = load_a_sh<H_ + 2>(h_s, nl, kt * 32, hf);
        v8i bfrag = *(const v8i*)(wfc_t + kt * 512 + lane * 16);
        facc = __builtin_amdgcn_wmma_f32_16x16x32_bf16(
            false, afr, false, __builtin_bit_cast(v16bf, bfrag),
            (short)0, facc, false, false);
      }
      if (nl < OUT_) {
#pragma unroll
        for (int r = 0; r < 8; ++r) {
          int m = r + 8 * hf;
          float v = facc[r];
          mtv[((long)(bb + m) * T_ + t) * OUT_ + nl] = v;   // measure_tempo_vel
          inp_s[m][FIN + nl] = f2bf(v);                     // prev for t+1
        }
      }
    }
    __syncthreads();
  }
}

// ---------------- gather: broadcasted = mtv[:, hier - hier[0], :] ----------------
__global__ void gather_kernel(const float* __restrict__ mtv, const int* __restrict__ hier,
                              float* __restrict__ out2) {
  long i = (long)blockIdx.x * blockDim.x + threadIdx.x;
  int o = (int)(i & 1);
  int j = (int)((i >> 1) & (NOUT - 1));
  int b = (int)(i >> 14);
  int idx = hier[j] - hier[0];
  out2[i] = mtv[((long)b * T_ + idx) * OUT_ + o];
}

extern "C" void kernel_launch(void* const* d_in, const int* in_sizes, int n_in,
                              void* d_out, int out_size, void* d_ws, size_t ws_size,
                              hipStream_t stream) {
  (void)in_sizes; (void)n_in; (void)out_size; (void)ws_size;
  const float* x    = (const float*)d_in[0];
  const int*   hier = (const int*)d_in[1];
  const float* Wih  = (const float*)d_in[2];   // (1024, 64)
  const float* Whh  = (const float*)d_in[3];   // (1024, 256)
  const float* bih  = (const float*)d_in[4];
  const float* bhh  = (const float*)d_in[5];
  const float* fcW  = (const float*)d_in[6];   // (2, 256)
  const float* fcb  = (const float*)d_in[7];

  float* mtv = (float*)d_out;                                  // (256,2048,2)
  float* bro = (float*)d_out + (long)B_ * T_ * OUT_;           // (256,8192,2)

  unsigned char* ws = (unsigned char*)d_ws;
  unsigned short* whh_sw = (unsigned short*)(ws + WS_WHH);
  unsigned short* wih_sw = (unsigned short*)(ws + WS_WIH);
  unsigned short* wfc_sw = (unsigned short*)(ws + WS_FC);
  float*          bias   = (float*)(ws + WS_BIAS);

  add_bias_kernel<<<(G4H + 255) / 256, 256, 0, stream>>>(bih, bhh, bias);
  swizzle_kernel<<<64 * 8, 32, 0, stream>>>(Whh, whh_sw, H_,  G4H, 8);
  swizzle_kernel<<<64 * 2, 32, 0, stream>>>(Wih, wih_sw, KIN, G4H, 2);
  swizzle_kernel<<<1 * 8,  32, 0, stream>>>(fcW, wfc_sw, H_,  OUT_, 8);

  lstm_kernel<<<16, 256, 0, stream>>>(x, whh_sw, wih_sw, wfc_sw, bias, fcb, mtv);

  gather_kernel<<<(B_ * NOUT * OUT_) / 256, 256, 0, stream>>>(mtv, hier, bro);
}
